// MotionCorrectTest_4784593568349
// MI455X (gfx1250) — matrix operand order repaired and checked by reading, not compiled, not run on hardware
//
#include <hip/hip_runtime.h>
#include <hip/hip_bf16.h>
#include <math.h>

// Problem constants (from reference)
#define Hh   1024
#define Ww   1024
#define CSc  62      // crop start for Xc (900x900)
#define OFFc 72      // crop start for template (880x880)
#define TWc  880     // template width
#define Sc   21      // search positions per axis
#define MSc  10
#define EPSf 1e-8f
#define KPAD 928     // K padded to multiple of 32 (900 -> 928)

typedef __attribute__((ext_vector_type(16))) __bf16 v16bf;
typedef __attribute__((ext_vector_type(8)))  float  v8f;

// workspace layout (float offsets)
#define WS_SUMX 0        // sum of Xc (900x900)
#define WS_SUMT 1        // sum of tmpl (880x880)
#define WS_TVAR 2        // sum of t_zm^2
#define WS_XS   3
#define WS_YS   4
#define WS_NUM  8        // 32x32 = 1024 floats (WMMA GEMM output, atomically accumulated)
#define WS_VAR  1040     // 441 floats
#define WS_RS   1536     // 900*21 row window sums
#define WS_RSQ  20480    // 900*21 row window sums of squares

// ---------------- init: zero the atomic accumulators ----------------
__global__ void k_init(float* ws) {
  int i = blockIdx.x * 256 + threadIdx.x;
  if (i < 1040) ws[i] = 0.0f;
}

// ---------------- means of Xc and tmpl ----------------
__global__ void k_sums(const float* __restrict__ X, const float* __restrict__ T, float* ws) {
  __shared__ float s1[256], s2[256];
  int tid = threadIdx.x;
  long gid = (long)blockIdx.x * 256 + tid;
  long stride = (long)gridDim.x * 256;
  float a = 0.0f, b = 0.0f;
  for (long i = gid; i < 900L * 900L; i += stride) {
    int r = (int)(i / 900), c = (int)(i % 900);
    a += X[(CSc + r) * Ww + CSc + c];
  }
  for (long i = gid; i < 880L * 880L; i += stride) {
    int r = (int)(i / 880), c = (int)(i % 880);
    b += T[(OFFc + r) * Ww + OFFc + c];
  }
  s1[tid] = a; s2[tid] = b;
  __syncthreads();
  for (int o = 128; o > 0; o >>= 1) {
    if (tid < o) { s1[tid] += s1[tid + o]; s2[tid] += s2[tid + o]; }
    __syncthreads();
  }
  if (tid == 0) { atomicAdd(&ws[WS_SUMX], s1[0]); atomicAdd(&ws[WS_SUMT], s2[0]); }
}

// ---------------- t_var = sum((tmpl - mean)^2) ----------------
__global__ void k_tvar(const float* __restrict__ T, float* ws) {
  __shared__ float s1[256];
  int tid = threadIdx.x;
  long gid = (long)blockIdx.x * 256 + tid;
  long stride = (long)gridDim.x * 256;
  float tmean = ws[WS_SUMT] * (1.0f / 774400.0f);
  float a = 0.0f;
  for (long i = gid; i < 880L * 880L; i += stride) {
    int r = (int)(i / 880), c = (int)(i % 880);
    float v = T[(OFFc + r) * Ww + OFFc + c] - tmean;
    a += v * v;
  }
  s1[tid] = a;
  __syncthreads();
  for (int o = 128; o > 0; o >>= 1) {
    if (tid < o) s1[tid] += s1[tid + o];
    __syncthreads();
  }
  if (tid == 0) atomicAdd(&ws[WS_TVAR], s1[0]);
}

// ---------------- per-row sliding window sums over Xc (raw + squared) ----------------
// one wave32 per row; base reduction across lanes, then lane 0 slides 21 positions
__global__ void k_rowsums(const float* __restrict__ X, float* ws) {
  int lane = threadIdx.x & 31;
  int wv = (int)((blockIdx.x * blockDim.x + threadIdx.x) >> 5);
  if (wv >= 900) return;
  const float* row = X + (CSc + wv) * Ww + CSc;
  float s = 0.0f, sq = 0.0f;
  for (int c = lane; c < TWc; c += 32) { float v = row[c]; s += v; sq += v * v; }
  for (int o = 16; o > 0; o >>= 1) { s += __shfl_xor(s, o, 32); sq += __shfl_xor(sq, o, 32); }
  if (lane == 0) {
    float* rs = ws + WS_RS  + wv * Sc;
    float* rq = ws + WS_RSQ + wv * Sc;
    rs[0] = s; rq[0] = sq;
    for (int sy = 1; sy < Sc; sy++) {
      float vout = row[sy - 1], vin = row[sy - 1 + TWc];
      s  += vin - vout;
      sq += vin * vin - vout * vout;
      rs[sy] = s; rq[sy] = sq;
    }
  }
}

// ---------------- per-window variance from row sums ----------------
__global__ void k_var(float* ws) {
  int t = threadIdx.x;
  if (t >= Sc * Sc) return;
  int sx = t / Sc, sy = t % Sc;
  float ls = 0.0f, lq = 0.0f;
  for (int r = 0; r < TWc; r++) {
    ls += ws[WS_RS  + (sx + r) * Sc + sy];
    lq += ws[WS_RSQ + (sx + r) * Sc + sy];
  }
  float v = lq - ls * ls * (1.0f / ((float)TWc * (float)TWc)) + EPSf;
  ws[WS_VAR + t] = fmaxf(v, 0.0f);
}

// ---------------- WMMA correlation ----------------
// num[sx,sy] = sum_r ( A_r[32 x 928] * B_r[928 x 32] )  (Toeplitz B from template row r)
// one workgroup per r; 4 waves = the four 16x16 output tiles; bf16 inputs, f32 accum.
// A slab and B row are staged in LDS as bf16 so the K-loop is ds_load + v_wmma only.
__global__ void __launch_bounds__(128) k_corr(const float* __restrict__ X,
                                              const float* __restrict__ T,
                                              float* ws) {
  __shared__ __bf16 ab[32][KPAD];   // zero-mean Xc slab rows r..r+31, cols 0..927 (59.4 KB)
  __shared__ __bf16 tzb[2][1008];   // padded zero-mean template row, 2 parity copies (4 KB)
  int r = blockIdx.x;
  int tid = threadIdx.x;
  float tmean = ws[WS_SUMT] * (1.0f / 774400.0f);
  float xmean = ws[WS_SUMX] * (1.0f / 810000.0f);

  // stage B: tzb[p][i] = t_zm[i + p - 32], zero outside [0,880)
  const float* trow = T + (OFFc + r) * Ww + OFFc;
  for (int i = tid; i < 1008; i += 128) {
    int i0 = i - 32, i1 = i - 31;
    float v0 = (i0 >= 0 && i0 < TWc) ? (trow[i0] - tmean) : 0.0f;
    float v1 = (i1 >= 0 && i1 < TWc) ? (trow[i1] - tmean) : 0.0f;
    tzb[0][i] = (__bf16)v0;
    tzb[1][i] = (__bf16)v1;
  }
  // stage A: ab[row][k] = Xc[r+row][k] - xmean  (X rows 62+r..93+r, cols 62..989; all in-bounds)
  for (int p = tid; p < 32 * (KPAD / 2); p += 128) {
    int row = p / (KPAD / 2), cp = p % (KPAD / 2);
    const float* src = X + (CSc + r + row) * Ww + CSc + 2 * cp;
    ab[row][2 * cp]     = (__bf16)(src[0] - xmean);
    ab[row][2 * cp + 1] = (__bf16)(src[1] - xmean);
  }
  __syncthreads();

  int w = tid >> 5, lane = tid & 31;
  int m = w & 1;          // sx tile
  int n = w >> 1;         // sy tile
  // A k order used here (contiguous, not AMD's table order): lanes 0-15 hold K = 0..15,
  //                                        lanes 16-31 hold K = 16..31
  const __bf16* ap = &ab[16 * m + (lane & 15)][0];
  int abase = (lane < 16) ? 0 : 16;
  // B k order used here (contiguous, matches A): lanes 0-15 hold K=0..15, lanes 16-31 K=16..31, N=lane&15
  int nn  = 16 * n + (lane & 15);
  int kbB = (lane < 16) ? 0 : 16;

  v8f acc = {};
#pragma unroll 4
  for (int k0 = 0; k0 < KPAD; k0 += 32) {
    v16bf a, b;
    const __bf16* a0 = ap + k0 + abase;          // 16B-aligned groups of 8 bf16
#pragma unroll
    for (int t2 = 0; t2 < 8; ++t2) {
      a[t2]     = a0[t2];
      a[t2 + 8] = a0[8 + t2];
    }
    int bb  = 32 + k0 + kbB - nn;                // virtual index into t_zm window [1, 944]
    int par = bb & 1;
    const __bf16* bp = &tzb[par][bb - par];      // dword-aligned start
#pragma unroll
    for (int t2 = 0; t2 < 16; ++t2) b[t2] = bp[t2];
    acc = __builtin_amdgcn_wmma_f32_16x16x32_bf16(false, a, false, b, (short)0, acc,
                                                  false, false);
  }
  // C layout: element v -> (M = v + (lane<16?0:8), N = lane&15)
  int osx = 16 * m + ((lane < 16) ? 0 : 8);
  int osy = 16 * n + (lane & 15);
#pragma unroll
  for (int v = 0; v < 8; ++v)
    atomicAdd(&ws[WS_NUM + (osx + v) * 32 + osy], acc[v]);
}

// ---------------- ncc, argmax, sub-pixel peak ----------------
__global__ void __launch_bounds__(512) k_peak(float* ws, float* out) {
  __shared__ float sncc[Sc * Sc];
  __shared__ float sval[512];
  __shared__ int   sidx[512];
  int t = threadIdx.x;
  float v = -INFINITY;
  if (t < Sc * Sc) {
    int sx = t / Sc, sy = t % Sc;
    float num  = ws[WS_NUM + sx * 32 + sy];
    float tvar = ws[WS_TVAR] + EPSf;
    float var  = ws[WS_VAR + t];
    float ncc  = num / sqrtf(tvar * var);
    if (isnan(ncc)) ncc = 0.0f;
    sncc[t] = ncc;
    v = ncc;
  }
  sval[t] = v; sidx[t] = t;
  __syncthreads();
  for (int o = 256; o > 0; o >>= 1) {
    if (t < o) {
      float v2 = sval[t + o]; int i2 = sidx[t + o];
      if (v2 > sval[t] || (v2 == sval[t] && i2 < sidx[t])) { sval[t] = v2; sidx[t] = i2; }
    }
    __syncthreads();
  }
  if (t == 0) {
    int idx = sidx[0];
    int sx = idx / Sc, sy = idx % Sc;
    auto at = [&](int rr, int cc) {
      rr = min(max(rr, 0), Sc - 1);
      cc = min(max(cc, 0), Sc - 1);
      return logf(sncc[rr * Sc + cc]);
    };
    float l4  = 4.0f * at(sx, sy);
    float lxm = at(sx - 1, sy), lxp = at(sx + 1, sy);
    float lym = at(sx, sy - 1), lyp = at(sx, sy + 1);
    float xs = -((float)sx - (float)MSc) - (lxm - lxp) / (2.0f * lxm - l4 + 2.0f * lxp);
    float ys = -((float)sy - (float)MSc) - (lym - lyp) / (2.0f * lym - l4 + 2.0f * lyp);
    ws[WS_XS] = xs; ws[WS_YS] = ys;
    out[Hh * Ww]     = xs;
    out[Hh * Ww + 1] = ys;
  }
}

// ---------------- bilinear resample + transpose-flatten ----------------
__global__ void k_resample(const float* __restrict__ X, const float* __restrict__ ws,
                           float* __restrict__ out) {
  int gid = blockIdx.x * 256 + threadIdx.x;
  if (gid >= Hh * Ww) return;
  int r = gid / Ww, c = gid % Ww;
  float xs = ws[WS_XS], ys = ws[WS_YS];
  float rr = (float)r - xs, cc = (float)c - ys;
  float r0 = floorf(rr), c0 = floorf(cc);
  float wr = rr - r0, wc = cc - c0;
  int r0i = (int)r0, c0i = (int)c0;
  auto samp = [&](int ri, int ci) {
    return (ri >= 0 && ri < Hh && ci >= 0 && ci < Ww) ? X[ri * Ww + ci] : 0.0f;
  };
  float o = (1.0f - wr) * (1.0f - wc) * samp(r0i, c0i)
          + (1.0f - wr) * wc          * samp(r0i, c0i + 1)
          + wr          * (1.0f - wc) * samp(r0i + 1, c0i)
          + wr          * wc          * samp(r0i + 1, c0i + 1);
  out[c * Hh + r] = o;   // flat = out.T.reshape(-1)
}

extern "C" void kernel_launch(void* const* d_in, const int* in_sizes, int n_in,
                              void* d_out, int out_size, void* d_ws, size_t ws_size,
                              hipStream_t stream) {
  const float* X = (const float*)d_in[0];        // (1,1024,1024,1) f32
  const float* T = (const float*)d_in[1];        // (1024,1024) f32
  float* out = (float*)d_out;                    // 1024*1024 + 2 floats
  float* ws  = (float*)d_ws;

  k_init    <<<5,    256, 0, stream>>>(ws);
  k_sums    <<<256,  256, 0, stream>>>(X, T, ws);
  k_tvar    <<<256,  256, 0, stream>>>(T, ws);
  k_rowsums <<<113,  256, 0, stream>>>(X, ws);
  k_var     <<<1,    512, 0, stream>>>(ws);
  k_corr    <<<880,  128, 0, stream>>>(X, T, ws);
  k_peak    <<<1,    512, 0, stream>>>(ws, out);
  k_resample<<<4096, 256, 0, stream>>>(X, ws, out);
}
